// GNNModelClassification_25761213841426
// MI455X (gfx1250) — compile-verified, matrix operand order checked by reading, not run on hardware
//
#include <hip/hip_runtime.h>
#include <math.h>

typedef __attribute__((ext_vector_type(16))) _Float16 v16h;
typedef __attribute__((ext_vector_type(8)))  float    v8f;

// K index held by vector element e (0..15) for half h (0: lanes 0-15, 1: lanes 16-31)
// per CDNA5 ISA 7.12.2 "16-bit A-Matrix 16x32" layout (B mirrored with N<->M).
__device__ __forceinline__ int wmma_kmap(int e, int h) {
  int j = e >> 1, r = e & 1;
  return (j < 4) ? (h * 8 + 2 * j + r) : (16 + h * 8 + 2 * (j - 4) + r);
}

// Pre-pack the 4 weight matrices into per-lane WMMA B fragments (f16, K zero-padded
// DIN..31). One wave per fragment; fragment index = m * (DOUT/16) + ct.
// wpack[frag * 32 + lane] = 16 halfs (32 B) in register-element order.
template<int DIN, int DOUT>
__global__ void pack_w_kernel(const float* __restrict__ w0, const float* __restrict__ w1,
                              const float* __restrict__ w2, const float* __restrict__ w3,
                              v16h* __restrict__ wpack) {
  constexpr int NCT = DOUT / 16;
  const int lane = threadIdx.x & 31;
  const int frag = blockIdx.x;                 // m*NCT + ct
  const int m = frag / NCT, ct = frag % NCT;
  const float* W[4] = {w0, w1, w2, w3};
  const int h = lane >> 4;
  const int col = ct * 16 + (lane & 15);
  v16h b;
#pragma unroll
  for (int e = 0; e < 16; ++e) {
    const int k  = wmma_kmap(e, h);
    const int kc = (k < DIN) ? k : (DIN - 1);             // clamped, unconditional load
    const float wv = W[m][(size_t)kc * DOUT + col];
    b[e] = (k < DIN) ? (_Float16)wv : (_Float16)0.0f;     // branch-free select
  }
  wpack[(size_t)frag * 32 + lane] = b;
}

// Fused 4-way node linear over FULL 16-node tiles only:
//   o_i = x @ w_i + b_i (i = 0..3) via v_wmma_f32_16x16x32_f16.
// B fragments come pre-packed from wpack (one 32 B vector load per lane per fragment).
// Tail nodes (n % 16) are handled by lin4_tail_kernel.
template<int DIN, int DOUT>
__global__ void lin4_wmma_kernel(const float* __restrict__ x,
                                 const v16h* __restrict__ wpack,
                                 const float* __restrict__ b0, float* __restrict__ o0,
                                 const float* __restrict__ b1, float* __restrict__ o1,
                                 const float* __restrict__ b2, float* __restrict__ o2,
                                 const float* __restrict__ b3, float* __restrict__ o3,
                                 int n_tiles) {
  static_assert(DIN <= 32 && (DOUT % 16) == 0, "K padded to 32, N multiple of 16");
  constexpr int NCT = DOUT / 16;
  const int lane = threadIdx.x & 31;
  const int wave = threadIdx.x >> 5;
  const int tile = blockIdx.x * (blockDim.x >> 5) + wave;
  if (tile >= n_tiles) return;        // wave-uniform exit: WMMA waves keep EXEC all-ones

  const int h = lane >> 4;            // which K-half this lane holds
  const int r = lane & 15;            // A: row M; B/D: column N
  const int row = tile * 16 + r;      // always valid (full tiles only)

  // ---- A tile: 16 nodes x 32 K (f16), zero-padded K >= DIN ----
  v16h a;
#pragma unroll
  for (int e = 0; e < 16; ++e) {
    const int k  = wmma_kmap(e, h);
    const int kc = (k < DIN) ? k : (DIN - 1);             // clamped, unconditional load
    const float xv = x[(size_t)row * DIN + kc];
    a[e] = (k < DIN) ? (_Float16)xv : (_Float16)0.0f;     // branch-free select
  }

  const float* Bv[4] = {b0, b1, b2, b3};
  float*       O[4]  = {o0, o1, o2, o3};

#pragma unroll
  for (int m = 0; m < 4; ++m) {
#pragma unroll
    for (int ct = 0; ct < NCT; ++ct) {
      const int col = ct * 16 + r;
      const v16h b = wpack[(size_t)(m * NCT + ct) * 32 + lane];  // 2x global_load_b128
      v8f c = {};
      c = __builtin_amdgcn_wmma_f32_16x16x32_f16(false, a, false, b, (short)0, c, false, false);
      const float bias = Bv[m][col];
      // D layout: lanes 0-15 -> M=e, N=lane; lanes 16-31 -> M=e+8, N=lane-16
      float* obase = O[m] + (size_t)(tile * 16 + 8 * h) * DOUT + col;
#pragma unroll
      for (int e = 0; e < 8; ++e) obase[(size_t)e * DOUT] = c[e] + bias;
    }
  }
}

// Scalar fallback for the n % 16 tail nodes (launched only when tail > 0).
// Thread id -> (m, node offset in tail, col).
template<int DIN, int DOUT>
__global__ void lin4_tail_kernel(const float* __restrict__ x,
                                 const float* __restrict__ w0, const float* __restrict__ b0, float* __restrict__ o0,
                                 const float* __restrict__ w1, const float* __restrict__ b1, float* __restrict__ o1,
                                 const float* __restrict__ w2, const float* __restrict__ b2, float* __restrict__ o2,
                                 const float* __restrict__ w3, const float* __restrict__ b3, float* __restrict__ o3,
                                 int node0, int n_nodes) {
  const float* W[4]  = {w0, w1, w2, w3};
  const float* Bv[4] = {b0, b1, b2, b3};
  float*       O[4]  = {o0, o1, o2, o3};
  const int tail = n_nodes - node0;
  int t = blockIdx.x * blockDim.x + threadIdx.x;
  const int total = 4 * tail * DOUT;
  if (t >= total) return;
  const int m    = t / (tail * DOUT);
  const int rem  = t % (tail * DOUT);
  const int node = node0 + rem / DOUT;
  const int col  = rem % DOUT;
  float acc = Bv[m][col];
  for (int k = 0; k < DIN; ++k)
    acc += x[(size_t)node * DIN + k] * W[m][(size_t)k * DOUT + col];
  O[m][(size_t)node * DOUT + col] = acc;
}

// ---------------- edge passes (scatter softmax + aggregate) ----------------

// Pass 1: score[e] = <q[dst], k[src]> / sqrt(D); segment-max via order-preserving uint atomicMax.
template<int D>
__global__ void edge_score_kernel(const float* __restrict__ q, const float* __restrict__ k,
                                  const int* __restrict__ src, const int* __restrict__ dst,
                                  float* __restrict__ escore, unsigned int* __restrict__ m_u,
                                  int n_edges) {
  constexpr int L = D / 4;                       // lanes per edge (4 or 8), aligned sub-groups
  size_t gid = (size_t)blockIdx.x * blockDim.x + threadIdx.x;
  int edge = (int)(gid / L);
  int sub  = (int)(gid % L);
  if (edge >= n_edges) return;
  const int sn = src[edge], dn = dst[edge];
  const float4 qv = ((const float4*)(q + (size_t)dn * D))[sub];
  const float4 kv = ((const float4*)(k + (size_t)sn * D))[sub];
  float p = qv.x * kv.x + qv.y * kv.y + qv.z * kv.z + qv.w * kv.w;
#pragma unroll
  for (int off = 1; off < L; off <<= 1) p += __shfl_xor(p, off, 32);
  if (sub == 0) {
    const float invsq = (D == 16) ? 0.25f : 0.17677669529663688f;  // 1/sqrt(D)
    const float score = p * invsq;
    escore[edge] = score;
    unsigned int u = __float_as_uint(score);
    u = (u & 0x80000000u) ? ~u : (u | 0x80000000u); // order-preserving float->uint
    atomicMax(&m_u[dn], u);
  }
}

// Pass 2: e = exp(score - m[dst]); segment-sum via atomicAdd. Overwrites escore with e.
__global__ void edge_exp_kernel(float* __restrict__ escore,
                                const int* __restrict__ dst,
                                const unsigned int* __restrict__ m_u,
                                float* __restrict__ ssum, int n_edges) {
  int e = blockIdx.x * blockDim.x + threadIdx.x;
  if (e >= n_edges) return;
  const int dn = dst[e];
  const unsigned int u = m_u[dn];
  const float m = (u & 0x80000000u) ? __uint_as_float(u ^ 0x80000000u) : __uint_as_float(~u);
  const float ev = __expf(escore[e] - m);
  escore[e] = ev;
  atomicAdd(&ssum[dn], ev);
}

// Pass 3: out[dst] += alpha * v[src], alpha = e / (s[dst]+1e-16). out pre-seeded with skip term.
template<int D>
__global__ void edge_agg_kernel(const float* __restrict__ v, const float* __restrict__ escore,
                                const float* __restrict__ ssum,
                                const int* __restrict__ src, const int* __restrict__ dst,
                                float* __restrict__ out, int n_edges) {
  constexpr int L = D / 4;
  size_t gid = (size_t)blockIdx.x * blockDim.x + threadIdx.x;
  int edge = (int)(gid / L);
  int sub  = (int)(gid % L);
  if (edge >= n_edges) return;
  const int sn = src[edge], dn = dst[edge];
  const float alpha = escore[edge] / (ssum[dn] + 1e-16f);
  const float4 vv = ((const float4*)(v + (size_t)sn * D))[sub];
  float* o = out + (size_t)dn * D + sub * 4;
  atomicAdd(o + 0, alpha * vv.x);
  atomicAdd(o + 1, alpha * vv.y);
  atomicAdd(o + 2, alpha * vv.z);
  atomicAdd(o + 3, alpha * vv.w);
}

__global__ void relu_kernel(float* __restrict__ p, size_t n) {
  size_t i = (size_t)blockIdx.x * blockDim.x + threadIdx.x;
  if (i < n) p[i] = fmaxf(p[i], 0.0f);
}

__global__ void zero_kernel(float* __restrict__ p, size_t n) {
  size_t i = (size_t)blockIdx.x * blockDim.x + threadIdx.x;
  if (i < n) p[i] = 0.0f;
}

// Mean-pool accumulation: 8 lanes per node, float4 per lane.
__global__ void pool_kernel(const float* __restrict__ h, const int* __restrict__ batch,
                            float* __restrict__ summed, float* __restrict__ counts, int n_nodes) {
  size_t gid = (size_t)blockIdx.x * blockDim.x + threadIdx.x;
  int node = (int)(gid / 8);
  int sub  = (int)(gid % 8);
  if (node >= n_nodes) return;
  const int g = batch[node];
  const float4 hv = ((const float4*)(h + (size_t)node * 32))[sub];
  float* o = summed + (size_t)g * 32 + sub * 4;
  atomicAdd(o + 0, hv.x);
  atomicAdd(o + 1, hv.y);
  atomicAdd(o + 2, hv.z);
  atomicAdd(o + 3, hv.w);
  if (sub == 0) atomicAdd(&counts[g], 1.0f);
}

// out[g] = relu((summed[g]/max(count,1)) @ fc1_w + fc1_b), fc1_w: [32,2]
__global__ void fc_kernel(const float* __restrict__ summed, const float* __restrict__ counts,
                          const float* __restrict__ w, const float* __restrict__ b,
                          float* __restrict__ out, int n_graphs) {
  int g = blockIdx.x * blockDim.x + threadIdx.x;
  if (g >= n_graphs) return;
  const float inv = 1.0f / fmaxf(counts[g], 1.0f);
  float a0 = b[0], a1 = b[1];
#pragma unroll
  for (int k = 0; k < 32; ++k) {
    const float p = summed[(size_t)g * 32 + k] * inv;
    a0 += p * w[k * 2 + 0];
    a1 += p * w[k * 2 + 1];
  }
  out[g * 2 + 0] = fmaxf(a0, 0.0f);
  out[g * 2 + 1] = fmaxf(a1, 0.0f);
}

extern "C" void kernel_launch(void* const* d_in, const int* in_sizes, int n_in,
                              void* d_out, int out_size, void* d_ws, size_t ws_size,
                              hipStream_t stream) {
  const float* x     = (const float*)d_in[0];
  const int*   ei    = (const int*)d_in[1];
  const int*   batch = (const int*)d_in[2];
  const float* c1_wq = (const float*)d_in[3];  const float* c1_bq = (const float*)d_in[4];
  const float* c1_wk = (const float*)d_in[5];  const float* c1_bk = (const float*)d_in[6];
  const float* c1_wv = (const float*)d_in[7];  const float* c1_bv = (const float*)d_in[8];
  const float* c1_ws = (const float*)d_in[9];  const float* c1_bs = (const float*)d_in[10];
  const float* c2_wq = (const float*)d_in[11]; const float* c2_bq = (const float*)d_in[12];
  const float* c2_wk = (const float*)d_in[13]; const float* c2_bk = (const float*)d_in[14];
  const float* c2_wv = (const float*)d_in[15]; const float* c2_bv = (const float*)d_in[16];
  const float* c2_ws = (const float*)d_in[17]; const float* c2_bs = (const float*)d_in[18];
  const float* fc1_w = (const float*)d_in[19]; const float* fc1_b = (const float*)d_in[20];

  const int n_edges  = in_sizes[1] / 2;
  const int n_nodes  = in_sizes[2];
  const int n_graphs = out_size / 2;
  const int* src = ei;
  const int* dst = ei + n_edges;

  // ---- workspace carve-out ----
  char* ws = (char*)d_ws;
  size_t off = 0;
  auto carve = [&](size_t bytes) -> void* {
    void* p = ws + off;
    off = (off + bytes + 255) & ~(size_t)255;
    return p;
  };
  float* q1 = (float*)carve((size_t)n_nodes * 16 * 4);
  float* k1 = (float*)carve((size_t)n_nodes * 16 * 4);
  float* v1 = (float*)carve((size_t)n_nodes * 16 * 4);
  float* h1 = (float*)carve((size_t)n_nodes * 16 * 4);
  float* q2 = (float*)carve((size_t)n_nodes * 32 * 4);
  float* k2 = (float*)carve((size_t)n_nodes * 32 * 4);
  float* v2 = (float*)carve((size_t)n_nodes * 32 * 4);
  float* h2 = (float*)carve((size_t)n_nodes * 32 * 4);
  float* escore = (float*)carve((size_t)n_edges * 4);          // reused across layers
  unsigned int* m_u = (unsigned int*)carve((size_t)n_nodes * 4);
  float* ssum = (float*)carve((size_t)n_nodes * 4);
  float* summed = (float*)carve((size_t)n_graphs * 32 * 4);
  float* counts = (float*)carve((size_t)n_graphs * 4);
  v16h* wpack1 = (v16h*)carve((size_t)4 * 1 * 32 * sizeof(v16h));  // 4 fragments
  v16h* wpack2 = (v16h*)carve((size_t)4 * 2 * 32 * sizeof(v16h));  // 8 fragments
  (void)ws_size; (void)n_in;

  const int TB = 256;
  const int n_tiles    = n_nodes / 16;               // full WMMA tiles
  const int tail_node0 = n_tiles * 16;
  const int n_tail     = n_nodes - tail_node0;       // 0 for n_nodes = 100000
  const int lin_blocks = (n_tiles + 3) / 4;          // 4 waves / block

  auto grid1 = [&](size_t threads) { return (int)((threads + TB - 1) / TB); };

  // ===== Layer 1 (d = 16) =====
  pack_w_kernel<9, 16><<<4, 32, 0, stream>>>(c1_wq, c1_wk, c1_wv, c1_ws, wpack1);
  lin4_wmma_kernel<9, 16><<<lin_blocks, 128, 0, stream>>>(
      x, wpack1, c1_bq, q1, c1_bk, k1, c1_bv, v1, c1_bs, h1, n_tiles);
  if (n_tail > 0)
    lin4_tail_kernel<9, 16><<<grid1((size_t)4 * n_tail * 16), TB, 0, stream>>>(
        x, c1_wq, c1_bq, q1, c1_wk, c1_bk, k1, c1_wv, c1_bv, v1, c1_ws, c1_bs, h1,
        tail_node0, n_nodes);
  zero_kernel<<<grid1(n_nodes), TB, 0, stream>>>((float*)m_u, (size_t)n_nodes);
  zero_kernel<<<grid1(n_nodes), TB, 0, stream>>>(ssum, (size_t)n_nodes);
  edge_score_kernel<16><<<grid1((size_t)n_edges * 4), TB, 0, stream>>>(q1, k1, src, dst, escore, m_u, n_edges);
  edge_exp_kernel<<<grid1(n_edges), TB, 0, stream>>>(escore, dst, m_u, ssum, n_edges);
  edge_agg_kernel<16><<<grid1((size_t)n_edges * 4), TB, 0, stream>>>(v1, escore, ssum, src, dst, h1, n_edges);
  relu_kernel<<<grid1((size_t)n_nodes * 16), TB, 0, stream>>>(h1, (size_t)n_nodes * 16);

  // ===== Layer 2 (d = 32) =====
  pack_w_kernel<16, 32><<<8, 32, 0, stream>>>(c2_wq, c2_wk, c2_wv, c2_ws, wpack2);
  lin4_wmma_kernel<16, 32><<<lin_blocks, 128, 0, stream>>>(
      h1, wpack2, c2_bq, q2, c2_bk, k2, c2_bv, v2, c2_bs, h2, n_tiles);
  if (n_tail > 0)
    lin4_tail_kernel<16, 32><<<grid1((size_t)4 * n_tail * 32), TB, 0, stream>>>(
        h1, c2_wq, c2_bq, q2, c2_wk, c2_bk, k2, c2_wv, c2_bv, v2, c2_ws, c2_bs, h2,
        tail_node0, n_nodes);
  zero_kernel<<<grid1(n_nodes), TB, 0, stream>>>((float*)m_u, (size_t)n_nodes);
  zero_kernel<<<grid1(n_nodes), TB, 0, stream>>>(ssum, (size_t)n_nodes);
  edge_score_kernel<32><<<grid1((size_t)n_edges * 8), TB, 0, stream>>>(q2, k2, src, dst, escore, m_u, n_edges);
  edge_exp_kernel<<<grid1(n_edges), TB, 0, stream>>>(escore, dst, m_u, ssum, n_edges);
  edge_agg_kernel<32><<<grid1((size_t)n_edges * 8), TB, 0, stream>>>(v2, escore, ssum, src, dst, h2, n_edges);
  relu_kernel<<<grid1((size_t)n_nodes * 32), TB, 0, stream>>>(h2, (size_t)n_nodes * 32);

  // ===== Mean pool + FC =====
  zero_kernel<<<grid1((size_t)n_graphs * 32), TB, 0, stream>>>(summed, (size_t)n_graphs * 32);
  zero_kernel<<<grid1(n_graphs), TB, 0, stream>>>(counts, (size_t)n_graphs);
  pool_kernel<<<grid1((size_t)n_nodes * 8), TB, 0, stream>>>(h2, batch, summed, counts, n_nodes);
  fc_kernel<<<grid1(n_graphs), TB, 0, stream>>>(summed, counts, fc1_w, fc1_b, (float*)d_out, n_graphs);
}
